// Discriminator_63428077027879
// MI455X (gfx1250) — matrix operand order repaired and checked by reading, not compiled, not run on hardware
//
#include <hip/hip_runtime.h>
#include <hip/hip_bf16.h>

typedef __attribute__((ext_vector_type(16))) __bf16 v16bf;
typedef __attribute__((ext_vector_type(8)))  float  v8f;

#define E_DIM   16
#define HID     32
#define L_SEQ   64
#define LSTM_D  256

// ---------------- WMMA helpers (CDNA5 wave32, 16x16x32 bf16 -> f32) ----------------

__device__ __forceinline__ v8f wmma_bf16(v16bf a, v16bf b, v8f c) {
  // (neg_a, A, neg_b, B, c_mod, C, reuse_a, reuse_b)
  return __builtin_amdgcn_wmma_f32_16x16x32_bf16(false, a, false, b, (short)0, c, false, false);
}

// 16-bit A matrix 16x32 layout: lane l (row m=l&15), element e(0..15) holds
// K = (l>>4)*8 + (e>=8?16:0) + (e&7); pairs e=(2v,2v+1) -> consecutive K.
__device__ __forceinline__ int a_koff(int v, int hi) {
  return hi * 8 + ((v >= 4) ? 16 : 0) + 2 * (v & 3);
}

// B fragments pre-packed by pack_weight_kernel: 32 contiguous bytes per lane.
__device__ __forceinline__ v16bf load_b_frag(const __bf16* wp, int tn, int lane) {
  return *(const v16bf*)(wp + ((size_t)tn * 32 + lane) * 16);
}

// Build A fragment from a bf16 row-major [16 x stride] LDS buffer.
__device__ __forceinline__ v16bf load_a_lds(const __bf16* buf, int m, int stride,
                                            int kbase, int hi) {
  v16bf a;
#pragma unroll
  for (int v = 0; v < 8; v++) {
    int K = kbase + a_koff(v, hi);
    a[2 * v]     = buf[m * stride + K];
    a[2 * v + 1] = buf[m * stride + K + 1];
  }
  return a;
}

// Build A fragment from the gathered 480-feature vector:
// k<224 -> embedding[tok[k/16]][k%16], k>=224 -> lstm row.
// All (K,K+1) pairs stay inside one 16-float segment -> safe float2 loads.
__device__ __forceinline__ v16bf load_a_feat(const int* tok, const float* lrow,
                                             const float* emb, int hi, int t) {
  v16bf a;
#pragma unroll
  for (int v = 0; v < 8; v++) {
    int K = t * 32 + a_koff(v, hi);
    const float* src = (K < 224) ? (emb + tok[K >> 4] * E_DIM + (K & 15))
                                 : (lrow + (K - 224));
    float2 f = *(const float2*)src;
    a[2 * v]     = (__bf16)f.x;
    a[2 * v + 1] = (__bf16)f.y;
  }
  return a;
}

// D layout (16x16 f32): lane l holds col = 16n + (l&15), rows m = (l>=16?8:0)+g.
// Add bias, ReLU, scatter to bf16 row-major [16 x 128] LDS for the next layer's A.
__device__ __forceinline__ void relu_bias_store_h(const v8f* acc, const float* bias,
                                                  __bf16* hb, int m, int hi) {
#pragma unroll
  for (int n = 0; n < 8; n++) {
    float bs = bias[n * 16 + m];
#pragma unroll
    for (int g = 0; g < 8; g++) {
      float x = fmaxf(acc[n][g] + bs, 0.f);
      hb[(hi * 8 + g) * 128 + n * 16 + m] = (__bf16)x;
    }
  }
}

// ---------------- weight pre-pack: f32 (K x N row-major) -> bf16 B fragments ----------------
// packed[(((t*ntiles)+n)*32 + lane)*16 + e] = W[(t*32 + (lane>>4)*8 + (e>=8?16:0) + (e&7)) * N + n*16 + (lane&15)]

__global__ void pack_weight_kernel(const float* __restrict__ W, __bf16* __restrict__ out,
                                   int K, int N) {
  int tid = blockIdx.x * blockDim.x + threadIdx.x;
  if (tid >= K * N) return;
  int e = tid & 15;
  int lane = (tid >> 4) & 31;
  int rest = tid >> 9;
  int ntiles = N >> 4;
  int n = rest % ntiles;
  int t = rest / ntiles;
  int j = n * 16 + (lane & 15);
  int k = t * 32 + ((lane >> 4) << 3) + ((e >= 8) ? 16 : 0) + (e & 7);
  out[tid] = (__bf16)W[(size_t)k * N + j];
}

// ---------------- kernel 1: per-leaf MLP (480->128->32) + leaf_end override ----------------
// one wave per 16 leaves; 8 waves per block.

__global__ __launch_bounds__(256) void leaf_kernel(
    const int* __restrict__ trees, const float* __restrict__ lstm,
    const float* __restrict__ embedding,
    const float* __restrict__ leaf_end_W, const float* __restrict__ leaf_end_b,
    const __bf16* __restrict__ w1p, const float* __restrict__ b1,
    const __bf16* __restrict__ w2p, const float* __restrict__ b2,
    float* __restrict__ children) {
  __shared__ __bf16 s_h[8][16 * 128];
  const int w = threadIdx.x >> 5, lane = threadIdx.x & 31;
  const int m = lane & 15, hi = lane >> 4;
  const int rowbase = (blockIdx.x * 8 + w) * 16;
  const int r = rowbase + m;                 // leaf row this lane supplies A data for
  const int b = r >> 4, c = r & 15;          // C == 16
  const int* leaf = trees + ((size_t)b * 17 + 1 + c) * 17;
  int tok[14];
#pragma unroll
  for (int i = 0; i < 14; i++) tok[i] = leaf[2 + i];
  const int ptr = leaf[16];
  const float* lrow = lstm + ((size_t)b * L_SEQ + ptr) * LSTM_D;
  __bf16* hb = s_h[w];

  // layer 1: (16 x 480) @ (480 x 128): 15 k-tiles x 8 n-tiles WMMA
  v8f acc1[8] = {};
#pragma unroll
  for (int t = 0; t < 15; t++) {
    v16bf a = load_a_feat(tok, lrow, embedding, hi, t);
#pragma unroll
    for (int n = 0; n < 8; n++)
      acc1[n] = wmma_bf16(a, load_b_frag(w1p, t * 8 + n, lane), acc1[n]);
  }
  relu_bias_store_h(acc1, b1, hb, m, hi);
  __syncthreads();

  // layer 2: (16 x 128) @ (128 x 32): 4 k-tiles x 2 n-tiles
  v8f acc2[2] = {};
#pragma unroll
  for (int t = 0; t < 4; t++) {
    v16bf a = load_a_lds(hb, m, 128, t * 32, hi);
#pragma unroll
    for (int n = 0; n < 2; n++)
      acc2[n] = wmma_bf16(a, load_b_frag(w2p, t * 2 + n, lane), acc2[n]);
  }

  // leaf_end branch value: embedding[END_IDX==1] @ leaf_end_W + b -> constant per column
  float endv[2];
#pragma unroll
  for (int n = 0; n < 2; n++) {
    int col = n * 16 + m;
    float s = leaf_end_b[col];
#pragma unroll
    for (int i = 0; i < 16; i++) s += embedding[E_DIM + i] * leaf_end_W[i * HID + col];
    endv[n] = s;
  }

#pragma unroll
  for (int g = 0; g < 8; g++) {
    int r2 = rowbase + hi * 8 + g;
    int bb = r2 >> 4, cc = r2 & 15;
    int tok2 = trees[((size_t)bb * 17 + 1 + cc) * 17 + 2];
#pragma unroll
    for (int n = 0; n < 2; n++) {
      float x = fmaxf(acc2[n][g] + b2[n * 16 + m], 0.f);
      if (tok2 == 1) x = endv[n];
      children[(size_t)r2 * HID + n * 16 + m] = x;
    }
  }
}

// ---------------- kernel 2: node MLP + 15-step child scan + combine + ff + out ----------------
// one wave per 16 batch elements; 8 waves per block.

__global__ __launch_bounds__(256) void tree_kernel(
    const int* __restrict__ trees, const float* __restrict__ lstm,
    const float* __restrict__ embedding, const float* __restrict__ children,
    const __bf16* __restrict__ nf1p, const float* __restrict__ nf_b1,
    const __bf16* __restrict__ nf2p, const float* __restrict__ nf_b2,
    const __bf16* __restrict__ cs1p, const float* __restrict__ cs_b1,
    const __bf16* __restrict__ cs2p, const float* __restrict__ cs_b2,
    const __bf16* __restrict__ cb1p, const float* __restrict__ cb_b1,
    const __bf16* __restrict__ cb2p, const float* __restrict__ cb_b2,
    const __bf16* __restrict__ ff1p, const float* __restrict__ ff_b1,
    const __bf16* __restrict__ ff2p, const float* __restrict__ ff_b2,
    const float* __restrict__ ts_W, const float* __restrict__ ts_b,
    float* __restrict__ out) {
  __shared__ __bf16 s_cat[8][1024];   // [0..511]=carry/summary 16x32, [512..1023]=stage 16x32
  __shared__ __bf16 s_h[8][16 * 128]; // hidden staging; reused as f32 16x32 at the end
  const int w = threadIdx.x >> 5, lane = threadIdx.x & 31;
  const int m = lane & 15, hi = lane >> 4;
  const int bbase = (blockIdx.x * 8 + w) * 16;
  const int b = bbase + m;
  __bf16* cat = s_cat[w];
  __bf16* hb = s_h[w];

  // ---- node_tmp = MLP2(root features, nf) ----
  const int* root = trees + (size_t)b * 17 * 17;
  int tok[14];
#pragma unroll
  for (int i = 0; i < 14; i++) tok[i] = root[2 + i];
  const int ptr = root[16];
  const float* lrow = lstm + ((size_t)b * L_SEQ + ptr) * LSTM_D;

  v8f acc1[8] = {};
#pragma unroll
  for (int t = 0; t < 15; t++) {
    v16bf a = load_a_feat(tok, lrow, embedding, hi, t);
#pragma unroll
    for (int n = 0; n < 8; n++)
      acc1[n] = wmma_bf16(a, load_b_frag(nf1p, t * 8 + n, lane), acc1[n]);
  }
  relu_bias_store_h(acc1, nf_b1, hb, m, hi);
  __syncthreads();
  v8f nt[2] = {};
#pragma unroll
  for (int t = 0; t < 4; t++) {
    v16bf a = load_a_lds(hb, m, 128, t * 32, hi);
#pragma unroll
    for (int n = 0; n < 2; n++)
      nt[n] = wmma_bf16(a, load_b_frag(nf2p, t * 2 + n, lane), nt[n]);
  }
#pragma unroll
  for (int n = 0; n < 2; n++) {
    float bs = nf_b2[n * 16 + m];
#pragma unroll
    for (int g = 0; g < 8; g++) nt[n][g] = fmaxf(nt[n][g] + bs, 0.f);
  }
  __syncthreads();

  // ---- scan init: carry = children[:,0] ----
  for (int idx = lane; idx < 512; idx += 32) {
    int row = idx >> 5, col = idx & 31;
    cat[idx] = (__bf16)children[(((size_t)(bbase + row)) * 16 + 0) * 32 + col];
  }
  __syncthreads();

  // ---- 15 sequential scan steps: concat(carry, child) 64 -> 128 -> 32 ----
  for (int c = 1; c < 16; c++) {
    v8f h1[8] = {};
    v16bf a0 = load_a_lds(cat, m, 32, 0, hi);     // carry (first half of concat)
    v16bf a1;
#pragma unroll
    for (int v = 0; v < 8; v++) {                  // child (second half), f32 -> bf16
      int koff = a_koff(v, hi);
      float2 f = *(const float2*)(children +
                   (((size_t)(bbase + m)) * 16 + c) * 32 + koff);
      a1[2 * v]     = (__bf16)f.x;
      a1[2 * v + 1] = (__bf16)f.y;
    }
#pragma unroll
    for (int n = 0; n < 8; n++) {
      h1[n] = wmma_bf16(a0, load_b_frag(cs1p, 0 * 8 + n, lane), h1[n]);
      h1[n] = wmma_bf16(a1, load_b_frag(cs1p, 1 * 8 + n, lane), h1[n]);
    }
    relu_bias_store_h(h1, cs_b1, hb, m, hi);
    __syncthreads();
    v8f c2[2] = {};
#pragma unroll
    for (int t = 0; t < 4; t++) {
      v16bf a = load_a_lds(hb, m, 128, t * 32, hi);
#pragma unroll
      for (int n = 0; n < 2; n++)
        c2[n] = wmma_bf16(a, load_b_frag(cs2p, t * 2 + n, lane), c2[n]);
    }
    __syncthreads();
#pragma unroll
    for (int n = 0; n < 2; n++) {
      float bs = cs_b2[n * 16 + m];
#pragma unroll
      for (int g = 0; g < 8; g++)
        cat[(hi * 8 + g) * 32 + n * 16 + m] = (__bf16)fmaxf(c2[n][g] + bs, 0.f);
    }
    __syncthreads();
  }

  // ---- tree_sum = MLP2(concat(node_tmp, child_summary), cb) ----
#pragma unroll
  for (int n = 0; n < 2; n++)
#pragma unroll
    for (int g = 0; g < 8; g++)
      cat[512 + (hi * 8 + g) * 32 + n * 16 + m] = (__bf16)nt[n][g];
  __syncthreads();
  {
    v8f h1[8] = {};
    v16bf a0 = load_a_lds(cat + 512, m, 32, 0, hi);  // node_tmp first
    v16bf a1 = load_a_lds(cat, m, 32, 0, hi);        // summary second
#pragma unroll
    for (int n = 0; n < 8; n++) {
      h1[n] = wmma_bf16(a0, load_b_frag(cb1p, 0 * 8 + n, lane), h1[n]);
      h1[n] = wmma_bf16(a1, load_b_frag(cb1p, 1 * 8 + n, lane), h1[n]);
    }
    relu_bias_store_h(h1, cb_b1, hb, m, hi);
  }
  __syncthreads();
  v8f tsum[2] = {};
#pragma unroll
  for (int t = 0; t < 4; t++) {
    v16bf a = load_a_lds(hb, m, 128, t * 32, hi);
#pragma unroll
    for (int n = 0; n < 2; n++)
      tsum[n] = wmma_bf16(a, load_b_frag(cb2p, t * 2 + n, lane), tsum[n]);
  }
  __syncthreads();
#pragma unroll
  for (int n = 0; n < 2; n++) {
    float bs = cb_b2[n * 16 + m];
#pragma unroll
    for (int g = 0; g < 8; g++)
      cat[(hi * 8 + g) * 32 + n * 16 + m] = (__bf16)fmaxf(tsum[n][g] + bs, 0.f);
  }
  __syncthreads();

  // ---- ff: (x @ W1 + b1) @ W2 + b2   (no ReLU) ----
  v8f f1[2] = {};
  {
    v16bf a = load_a_lds(cat, m, 32, 0, hi);
#pragma unroll
    for (int n = 0; n < 2; n++)
      f1[n] = wmma_bf16(a, load_b_frag(ff1p, n, lane), f1[n]);
  }
  __syncthreads();
#pragma unroll
  for (int n = 0; n < 2; n++) {
    float bs = ff_b1[n * 16 + m];
#pragma unroll
    for (int g = 0; g < 8; g++)
      cat[512 + (hi * 8 + g) * 32 + n * 16 + m] = (__bf16)(f1[n][g] + bs);
  }
  __syncthreads();
  v8f f2[2] = {};
  {
    v16bf a = load_a_lds(cat + 512, m, 32, 0, hi);
#pragma unroll
    for (int n = 0; n < 2; n++)
      f2[n] = wmma_bf16(a, load_b_frag(ff2p, n, lane), f2[n]);
  }
  float* fbuf = (float*)hb;  // reuse hidden staging as 16x32 f32
#pragma unroll
  for (int n = 0; n < 2; n++) {
    float bs = ff_b2[n * 16 + m];
#pragma unroll
    for (int g = 0; g < 8; g++)
      fbuf[(hi * 8 + g) * 32 + n * 16 + m] = f2[n][g] + bs;
  }
  __syncthreads();

  // ---- out = ff @ ts_W + ts_b ----
  if (lane < 16) {
    float s = ts_b[0];
#pragma unroll
    for (int j = 0; j < 32; j++) s += fbuf[lane * 32 + j] * ts_W[j];
    out[bbase + lane] = s;
  }
}

// ---------------- host launch ----------------

extern "C" void kernel_launch(void* const* d_in, const int* in_sizes, int n_in,
                              void* d_out, int out_size, void* d_ws, size_t ws_size,
                              hipStream_t stream) {
  (void)in_sizes; (void)n_in; (void)out_size; (void)ws_size;
  const int*   trees      = (const int*)d_in[0];
  const float* lstm       = (const float*)d_in[1];
  // d_in[2] first_notes: unused by reference; d_in[3] train: unused
  const float* embedding  = (const float*)d_in[4];
  const float* leaf_end_W = (const float*)d_in[5];
  const float* leaf_end_b = (const float*)d_in[6];
  const float* lc_W1 = (const float*)d_in[7];
  const float* lc_b1 = (const float*)d_in[8];
  const float* lc_W2 = (const float*)d_in[9];
  const float* lc_b2 = (const float*)d_in[10];
  const float* cs_W1 = (const float*)d_in[11];
  const float* cs_b1 = (const float*)d_in[12];
  const float* cs_W2 = (const float*)d_in[13];
  const float* cs_b2 = (const float*)d_in[14];
  const float* nf_W1 = (const float*)d_in[15];
  const float* nf_b1 = (const float*)d_in[16];
  const float* nf_W2 = (const float*)d_in[17];
  const float* nf_b2 = (const float*)d_in[18];
  const float* cb_W1 = (const float*)d_in[19];
  const float* cb_b1 = (const float*)d_in[20];
  const float* cb_W2 = (const float*)d_in[21];
  const float* cb_b2 = (const float*)d_in[22];
  const float* ff_W1 = (const float*)d_in[23];
  const float* ff_b1 = (const float*)d_in[24];
  const float* ff_W2 = (const float*)d_in[25];
  const float* ff_b2 = (const float*)d_in[26];
  const float* ts_W  = (const float*)d_in[27];
  const float* ts_b  = (const float*)d_in[28];
  float* out = (float*)d_out;

  char* ws = (char*)d_ws;
  float* children = (float*)ws;                      // 8192*16*32 f32 = 16 MiB
  __bf16* wp = (__bf16*)(ws + (size_t)16 * 1024 * 1024);
  __bf16* lc1p = wp + 0;         // 480x128
  __bf16* lc2p = wp + 61440;     // 128x32
  __bf16* cs1p = wp + 65536;     // 64x128
  __bf16* cs2p = wp + 73728;     // 128x32
  __bf16* nf1p = wp + 77824;     // 480x128
  __bf16* nf2p = wp + 139264;    // 128x32
  __bf16* cb1p = wp + 143360;    // 64x128
  __bf16* cb2p = wp + 151552;    // 128x32
  __bf16* ff1p = wp + 155648;    // 32x32
  __bf16* ff2p = wp + 156672;    // 32x32

  auto pack = [&](const float* W, __bf16* o, int K, int N) {
    int total = K * N;
    pack_weight_kernel<<<(total + 255) / 256, 256, 0, stream>>>(W, o, K, N);
  };
  pack(lc_W1, lc1p, 480, 128);
  pack(lc_W2, lc2p, 128, 32);
  pack(cs_W1, cs1p, 64, 128);
  pack(cs_W2, cs2p, 128, 32);
  pack(nf_W1, nf1p, 480, 128);
  pack(nf_W2, nf2p, 128, 32);
  pack(cb_W1, cb1p, 64, 128);
  pack(cb_W2, cb2p, 128, 32);
  pack(ff_W1, ff1p, 32, 32);
  pack(ff_W2, ff2p, 32, 32);

  // 8192*16 leaves / (16 rows * 8 waves) = 1024 blocks
  leaf_kernel<<<1024, 256, 0, stream>>>(trees, lstm, embedding, leaf_end_W, leaf_end_b,
                                        lc1p, lc_b1, lc2p, lc_b2, children);
  // 8192 batch / (16 rows * 8 waves) = 64 blocks
  tree_kernel<<<64, 256, 0, stream>>>(trees, lstm, embedding, children,
                                      nf1p, nf_b1, nf2p, nf_b2,
                                      cs1p, cs_b1, cs2p, cs_b2,
                                      cb1p, cb_b1, cb2p, cb_b2,
                                      ff1p, ff_b1, ff2p, ff_b2,
                                      ts_W, ts_b, out);
}
